// SubGraph_89962384982779
// MI455X (gfx1250) — matrix operand order repaired and checked by reading, not compiled, not run on hardware
//
#include <hip/hip_runtime.h>

#define NROWS 1000000
#define NCLUS 50000

typedef __attribute__((ext_vector_type(16))) __bf16 v16bf;
typedef __attribute__((ext_vector_type(8)))  float  v8f;

union FragBF { unsigned int u[8]; v16bf v; };

__device__ __forceinline__ unsigned short f2bf(float f) {
  unsigned int u = __float_as_uint(f);
  u += 0x7FFFu + ((u >> 16) & 1u);            // round-to-nearest-even
  return (unsigned short)(u >> 16);
}

// monotone float -> u32 map (for signed segment-max via atomicMax on u32)
__device__ __forceinline__ unsigned int enc_f(float f) {
  unsigned int b = __float_as_uint(f);
  return b ^ ((unsigned int)((int)b >> 31) | 0x80000000u);
}
__device__ __forceinline__ float dec_u(unsigned int u) {
  if (u == 0u) return 0.0f;                   // empty segment marker
  unsigned int b = (u & 0x80000000u) ? (u ^ 0x80000000u) : ~u;
  return __uint_as_float(b);
}

// A-fragment loader: 16x32 bf16 tile, layout per CDNA5 ISA 7.12.2
// lane 0-15: M=lane, K groups {0..7,16..23}; lane 16-31: M=lane-16, K groups {8..15,24..31}
__device__ __forceinline__ v16bf load_a_frag(const unsigned short* base, int rowStride,
                                             int kb, int lane) {
  FragBF a;
  const int m = lane & 15, hi = lane >> 4;
#pragma unroll
  for (int v = 0; v < 8; ++v) {
    int k = kb * 32 + ((v < 4) ? 0 : 16) + hi * 8 + (v & 3) * 2;
    a.u[v] = *(const unsigned int*)(base + m * rowStride + k);
  }
  return a.v;
}

// B-fragment loader: 32x16 bf16 from row-major W[K][64]; lane%16 = N column, K order same as A fragment
__device__ __forceinline__ v16bf load_b_frag(const unsigned short* wmat, int kb, int nb, int lane) {
  FragBF b;
  const int n = nb * 16 + (lane & 15), hi = lane >> 4;
#pragma unroll
  for (int v = 0; v < 8; ++v) {
    int k = kb * 32 + ((v < 4) ? 0 : 16) + hi * 8 + (v & 3) * 2;
    b.u[v] = (unsigned int)wmat[k * 64 + n] | ((unsigned int)wmat[(k + 1) * 64 + n] << 16);
  }
  return b.v;
}

// ---------------------------------------------------------------------------
// Fused VectorNet MLP layer: [x | gather(mprev)] @ W1 + b1 -> LN -> ReLU
//                            -> @ W2 + b2 -> LN -> ReLU -> y (bf16) + segmax
// One wave handles a 16-row tile, 8 waves per block.
// ---------------------------------------------------------------------------
template<int K1, int DIN, bool GATHER>
__global__ __launch_bounds__(256)
void mlp_layer_kernel(const float* __restrict__ xf,            // layer0 raw input [N,DIN]
                      const unsigned short* __restrict__ xbf,  // bf16 activations [N,64]
                      const float* __restrict__ mprev,         // [C,64] f32 cluster max
                      const int* __restrict__ cluster,
                      const unsigned short* __restrict__ w1bf, // [K1,64] bf16
                      const unsigned short* __restrict__ w2bf, // [64,64] bf16
                      const float* __restrict__ b1, const float* __restrict__ g1,
                      const float* __restrict__ be1,
                      const float* __restrict__ b2, const float* __restrict__ g2,
                      const float* __restrict__ be2,
                      unsigned short* __restrict__ yout,       // [N,64] bf16
                      unsigned int* __restrict__ mout,         // [C,64] raw f32 bits (>=0)
                      int n) {
  constexpr int KA = (K1 > 64) ? K1 : 64;
  __shared__ unsigned short sW1[K1 * 64];
  __shared__ unsigned short sW2[64 * 64];
  __shared__ unsigned short sA[8][16 * KA];
  __shared__ float sM[8][16 * 64];

  const int tid  = threadIdx.x;
  const int wid  = tid >> 5, lane = tid & 31;
  const int m    = lane & 15, hi = lane >> 4;
  const int row0 = (blockIdx.x * 8 + wid) * 16;

  for (int i = tid; i < K1 * 32; i += 256)
    ((unsigned int*)sW1)[i] = ((const unsigned int*)w1bf)[i];
  for (int i = tid; i < 64 * 32; i += 256)
    ((unsigned int*)sW2)[i] = ((const unsigned int*)w2bf)[i];

  // ---- stage input tile (bf16) into sA ----
  if (GATHER) {
#pragma unroll 4
    for (int j = 0; j < (16 * K1) / 32; ++j) {
      int e = j * 32 + lane;
      int i = e / K1, k = e % K1;
      int row = row0 + i;
      unsigned short hv = 0;
      if (row < n) {
        if (k < 64) hv = xbf[(size_t)row * 64 + k];
        else        hv = f2bf(mprev[(size_t)cluster[row] * 64 + (k - 64)]);
      }
      sA[wid][i * KA + k] = hv;
    }
  } else {
#pragma unroll 4
    for (int j = 0; j < (16 * K1) / 32; ++j) {
      int e = j * 32 + lane;
      int i = e / K1, k = e % K1;
      int row = row0 + i;
      float val = (row < n && k < DIN) ? xf[(size_t)row * DIN + k] : 0.0f;
      sA[wid][i * KA + k] = f2bf(val);
    }
  }
  __syncthreads();

  // ---- GEMM1: [16,K1] x [K1,64] via v_wmma_f32_16x16x32_bf16 ----
  v8f acc[4];
#pragma unroll
  for (int t = 0; t < 4; ++t) {
#pragma unroll
    for (int r = 0; r < 8; ++r) acc[t][r] = 0.0f;
  }
#pragma unroll
  for (int kb = 0; kb < K1 / 32; ++kb) {
    v16bf afrag = load_a_frag(&sA[wid][0], KA, kb, lane);
#pragma unroll
    for (int nb = 0; nb < 4; ++nb) {
      v16bf bfrag = load_b_frag(sW1, kb, nb, lane);
      acc[nb] = __builtin_amdgcn_wmma_f32_16x16x32_bf16(
          false, afrag, false, bfrag, (short)0, acc[nb], false, false);
    }
  }
  // spill C tiles (+bias) to LDS: VGPR r -> row r + hi*8, col nb*16 + (lane&15)
#pragma unroll
  for (int nb = 0; nb < 4; ++nb) {
    int col = nb * 16 + m;
    float bb = b1[col];
#pragma unroll
    for (int r = 0; r < 8; ++r)
      sM[wid][(r + hi * 8) * 64 + col] = acc[nb][r] + bb;
  }
  __syncthreads();

  // ---- LayerNorm1 + ReLU -> bf16 hidden back into sA ----
  {
    float vals[32];
    float s = 0.0f;
#pragma unroll
    for (int j = 0; j < 32; ++j) { vals[j] = sM[wid][m * 64 + hi * 32 + j]; s += vals[j]; }
    s += __shfl_xor(s, 16);
    float mu = s * (1.0f / 64.0f);
    float vv = 0.0f;
#pragma unroll
    for (int j = 0; j < 32; ++j) { float d = vals[j] - mu; vv += d * d; }
    vv += __shfl_xor(vv, 16);
    float rstd = rsqrtf(vv * (1.0f / 64.0f) + 1e-5f);
#pragma unroll
    for (int j = 0; j < 32; ++j) {
      int col = hi * 32 + j;
      float o = (vals[j] - mu) * rstd * g1[col] + be1[col];
      sA[wid][m * KA + col] = f2bf(fmaxf(o, 0.0f));
    }
  }
  __syncthreads();

  // ---- GEMM2: [16,64] x [64,64] ----
#pragma unroll
  for (int t = 0; t < 4; ++t) {
#pragma unroll
    for (int r = 0; r < 8; ++r) acc[t][r] = 0.0f;
  }
#pragma unroll
  for (int kb = 0; kb < 2; ++kb) {
    v16bf afrag = load_a_frag(&sA[wid][0], KA, kb, lane);
#pragma unroll
    for (int nb = 0; nb < 4; ++nb) {
      v16bf bfrag = load_b_frag(sW2, kb, nb, lane);
      acc[nb] = __builtin_amdgcn_wmma_f32_16x16x32_bf16(
          false, afrag, false, bfrag, (short)0, acc[nb], false, false);
    }
  }
#pragma unroll
  for (int nb = 0; nb < 4; ++nb) {
    int col = nb * 16 + m;
    float bb = b2[col];
#pragma unroll
    for (int r = 0; r < 8; ++r)
      sM[wid][(r + hi * 8) * 64 + col] = acc[nb][r] + bb;
  }
  __syncthreads();

  // ---- LayerNorm2 + ReLU (result stays f32 in sM) ----
  {
    float vals[32];
    float s = 0.0f;
#pragma unroll
    for (int j = 0; j < 32; ++j) { vals[j] = sM[wid][m * 64 + hi * 32 + j]; s += vals[j]; }
    s += __shfl_xor(s, 16);
    float mu = s * (1.0f / 64.0f);
    float vv = 0.0f;
#pragma unroll
    for (int j = 0; j < 32; ++j) { float d = vals[j] - mu; vv += d * d; }
    vv += __shfl_xor(vv, 16);
    float rstd = rsqrtf(vv * (1.0f / 64.0f) + 1e-5f);
#pragma unroll
    for (int j = 0; j < 32; ++j) {
      int col = hi * 32 + j;
      float o = (vals[j] - mu) * rstd * g2[col] + be2[col];
      sM[wid][m * 64 + col] = fmaxf(o, 0.0f);
    }
  }
  __syncthreads();

  // ---- coalesced bf16 store of y ----
#pragma unroll 4
  for (int j = 0; j < 32; ++j) {
    int e = j * 32 + lane;
    int i = e >> 6, c = e & 63;
    int row = row0 + i;
    if (row < n) yout[(size_t)row * 64 + c] = f2bf(sM[wid][e]);
  }

  // ---- segment-max: run-length compress (cluster sorted) then atomics ----
  {
    const int c0 = lane, c1 = lane + 32;
    int curc = -1;
    float mx0 = 0.0f, mx1 = 0.0f;          // values are >= 0 (post-ReLU)
    for (int i = 0; i < 16; ++i) {
      int row = row0 + i;
      if (row >= n) break;
      int cid = cluster[row];
      if (cid != curc) {
        if (curc >= 0) {
          atomicMax(&mout[(size_t)curc * 64 + c0], __float_as_uint(mx0));
          atomicMax(&mout[(size_t)curc * 64 + c1], __float_as_uint(mx1));
        }
        curc = cid; mx0 = 0.0f; mx1 = 0.0f;
      }
      mx0 = fmaxf(mx0, sM[wid][i * 64 + c0]);
      mx1 = fmaxf(mx1, sM[wid][i * 64 + c1]);
    }
    if (curc >= 0) {
      atomicMax(&mout[(size_t)curc * 64 + c0], __float_as_uint(mx0));
      atomicMax(&mout[(size_t)curc * 64 + c1], __float_as_uint(mx1));
    }
  }
}

// ---------------------------------------------------------------------------
// Final: [y2 | gather(m2)] @ lin_w + lin_b -> signed segment-max (encoded u32)
// ---------------------------------------------------------------------------
__global__ __launch_bounds__(256)
void final_linear_kernel(const unsigned short* __restrict__ xbf,
                         const float* __restrict__ mprev,
                         const int* __restrict__ cluster,
                         const unsigned short* __restrict__ wbf,  // [128,64]
                         const float* __restrict__ bias,
                         unsigned int* __restrict__ mout,         // [C,64] encoded
                         int n) {
  __shared__ unsigned short sW[128 * 64];
  __shared__ unsigned short sA[8][16 * 128];
  __shared__ unsigned int   sU[8][16 * 64];

  const int tid  = threadIdx.x;
  const int wid  = tid >> 5, lane = tid & 31;
  const int m    = lane & 15, hi = lane >> 4;
  const int row0 = (blockIdx.x * 8 + wid) * 16;

  for (int i = tid; i < 128 * 32; i += 256)
    ((unsigned int*)sW)[i] = ((const unsigned int*)wbf)[i];

#pragma unroll 4
  for (int j = 0; j < (16 * 128) / 32; ++j) {
    int e = j * 32 + lane;
    int i = e >> 7, k = e & 127;
    int row = row0 + i;
    unsigned short hv = 0;
    if (row < n) {
      if (k < 64) hv = xbf[(size_t)row * 64 + k];
      else        hv = f2bf(mprev[(size_t)cluster[row] * 64 + (k - 64)]);
    }
    sA[wid][i * 128 + k] = hv;
  }
  __syncthreads();

  v8f acc[4];
#pragma unroll
  for (int t = 0; t < 4; ++t) {
#pragma unroll
    for (int r = 0; r < 8; ++r) acc[t][r] = 0.0f;
  }
#pragma unroll
  for (int kb = 0; kb < 4; ++kb) {
    v16bf afrag = load_a_frag(&sA[wid][0], 128, kb, lane);
#pragma unroll
    for (int nb = 0; nb < 4; ++nb) {
      v16bf bfrag = load_b_frag(sW, kb, nb, lane);
      acc[nb] = __builtin_amdgcn_wmma_f32_16x16x32_bf16(
          false, afrag, false, bfrag, (short)0, acc[nb], false, false);
    }
  }
#pragma unroll
  for (int nb = 0; nb < 4; ++nb) {
    int col = nb * 16 + m;
    float bb = bias[col];
#pragma unroll
    for (int r = 0; r < 8; ++r)
      sU[wid][(r + hi * 8) * 64 + col] = enc_f(acc[nb][r] + bb);
  }
  __syncthreads();

  {
    const int c0 = lane, c1 = lane + 32;
    int curc = -1;
    unsigned int mx0 = 0u, mx1 = 0u;
    for (int i = 0; i < 16; ++i) {
      int row = row0 + i;
      if (row >= n) break;
      int cid = cluster[row];
      if (cid != curc) {
        if (curc >= 0) {
          atomicMax(&mout[(size_t)curc * 64 + c0], mx0);
          atomicMax(&mout[(size_t)curc * 64 + c1], mx1);
        }
        curc = cid; mx0 = 0u; mx1 = 0u;
      }
      unsigned int t0 = sU[wid][i * 64 + c0], t1 = sU[wid][i * 64 + c1];
      mx0 = (t0 > mx0) ? t0 : mx0;
      mx1 = (t1 > mx1) ? t1 : mx1;
    }
    if (curc >= 0) {
      atomicMax(&mout[(size_t)curc * 64 + c0], mx0);
      atomicMax(&mout[(size_t)curc * 64 + c1], mx1);
    }
  }
}

__global__ __launch_bounds__(256)
void norm_rows_kernel(const unsigned int* __restrict__ mf, float* __restrict__ out, int c) {
  const int wid = threadIdx.x >> 5, lane = threadIdx.x & 31;
  const int ci = blockIdx.x * 8 + wid;
  if (ci >= c) return;
  float v0 = dec_u(mf[(size_t)ci * 64 + lane]);
  float v1 = dec_u(mf[(size_t)ci * 64 + lane + 32]);
  float s = v0 * v0 + v1 * v1;
#pragma unroll
  for (int off = 16; off > 0; off >>= 1) s += __shfl_xor(s, off);
  float scale = 1.0f / fmaxf(sqrtf(s), 1e-12f);
  out[(size_t)ci * 64 + lane]      = v0 * scale;
  out[(size_t)ci * 64 + lane + 32] = v1 * scale;
}

__global__ void zero_u32_kernel(unsigned int* p, int count) {
  int i = blockIdx.x * 256 + threadIdx.x;
  if (i < count) p[i] = 0u;
}

__global__ void prep_weights_kernel(const float* w1_0, const float* w2_0,
                                    const float* w1_1, const float* w2_1,
                                    const float* w1_2, const float* w2_2,
                                    const float* linw,
                                    unsigned short* W1_0, unsigned short* W2_0,
                                    unsigned short* W1_1, unsigned short* W2_1,
                                    unsigned short* W1_2, unsigned short* W2_2,
                                    unsigned short* WLIN) {
  int t = threadIdx.x;
  for (int i = t; i < 32 * 64; i += 256) {       // pad K: 10 -> 32
    int k = i >> 6, nn = i & 63;
    W1_0[i] = (k < 10) ? f2bf(w1_0[k * 64 + nn]) : (unsigned short)0;
  }
  for (int i = t; i < 64 * 64; i += 256)  W2_0[i] = f2bf(w2_0[i]);
  for (int i = t; i < 128 * 64; i += 256) W1_1[i] = f2bf(w1_1[i]);
  for (int i = t; i < 64 * 64; i += 256)  W2_1[i] = f2bf(w2_1[i]);
  for (int i = t; i < 128 * 64; i += 256) W1_2[i] = f2bf(w1_2[i]);
  for (int i = t; i < 64 * 64; i += 256)  W2_2[i] = f2bf(w2_2[i]);
  for (int i = t; i < 128 * 64; i += 256) WLIN[i] = f2bf(linw[i]);
}

extern "C" void kernel_launch(void* const* d_in, const int* in_sizes, int n_in,
                              void* d_out, int out_size, void* d_ws, size_t ws_size,
                              hipStream_t stream) {
  (void)out_size; (void)ws_size;
  const int n = NROWS, C = NCLUS;

  const float* x = nullptr;
  const int* cluster = nullptr;
  const float *w1[3], *b1[3], *g1[3], *be1[3], *w2[3], *b2[3], *g2[3], *be2[3];
  const float *linw = nullptr, *linb = nullptr;

  if (n_in >= 28 && in_sizes[0] == n * 10) {
    // insertion order: x, cluster, [w1,b1,g1,be1,w2,b2,g2,be2]x3, lin_w, lin_b
    x = (const float*)d_in[0];
    cluster = (const int*)d_in[1];
    int p = 2;
    for (int l = 0; l < 3; ++l) {
      w1[l]  = (const float*)d_in[p + 0]; b1[l]  = (const float*)d_in[p + 1];
      g1[l]  = (const float*)d_in[p + 2]; be1[l] = (const float*)d_in[p + 3];
      w2[l]  = (const float*)d_in[p + 4]; b2[l]  = (const float*)d_in[p + 5];
      g2[l]  = (const float*)d_in[p + 6]; be2[l] = (const float*)d_in[p + 7];
      p += 8;
    }
    linw = (const float*)d_in[26];
    linb = (const float*)d_in[27];
  } else {
    // alphabetical pytree order: cluster, lin_b, lin_w, [b1,b2,be1,be2,g1,g2,w1,w2]x3, x
    cluster = (const int*)d_in[0];
    linb = (const float*)d_in[1];
    linw = (const float*)d_in[2];
    int p = 3;
    for (int l = 0; l < 3; ++l) {
      b1[l]  = (const float*)d_in[p + 0]; b2[l]  = (const float*)d_in[p + 1];
      be1[l] = (const float*)d_in[p + 2]; be2[l] = (const float*)d_in[p + 3];
      g1[l]  = (const float*)d_in[p + 4]; g2[l]  = (const float*)d_in[p + 5];
      w1[l]  = (const float*)d_in[p + 6]; w2[l]  = (const float*)d_in[p + 7];
      p += 8;
    }
    x = (const float*)d_in[27];
  }

  // workspace layout
  unsigned char* ws = (unsigned char*)d_ws;
  size_t off = 0;
  unsigned short* yA = (unsigned short*)(ws + off); off += (size_t)n * 64 * 2;
  unsigned short* yB = (unsigned short*)(ws + off); off += (size_t)n * 64 * 2;
  unsigned int* mA = (unsigned int*)(ws + off); off += (size_t)C * 64 * 4;
  unsigned int* mB = (unsigned int*)(ws + off); off += (size_t)C * 64 * 4;  // contiguous after mA
  unsigned short* W1_0 = (unsigned short*)(ws + off); off += 32 * 64 * 2;
  unsigned short* W2_0 = (unsigned short*)(ws + off); off += 64 * 64 * 2;
  unsigned short* W1_1 = (unsigned short*)(ws + off); off += 128 * 64 * 2;
  unsigned short* W2_1 = (unsigned short*)(ws + off); off += 64 * 64 * 2;
  unsigned short* W1_2 = (unsigned short*)(ws + off); off += 128 * 64 * 2;
  unsigned short* W2_2 = (unsigned short*)(ws + off); off += 64 * 64 * 2;
  unsigned short* WLIN = (unsigned short*)(ws + off); off += 128 * 64 * 2;

  const dim3 blk(256);
  const int mlpBlocks = (n + 127) / 128;

  prep_weights_kernel<<<1, blk, 0, stream>>>(w1[0], w2[0], w1[1], w2[1], w1[2], w2[2], linw,
                                             W1_0, W2_0, W1_1, W2_1, W1_2, W2_2, WLIN);
  zero_u32_kernel<<<(2 * C * 64 + 255) / 256, blk, 0, stream>>>(mA, 2 * C * 64);

  mlp_layer_kernel<32, 10, false><<<mlpBlocks, blk, 0, stream>>>(
      x, nullptr, nullptr, cluster, W1_0, W2_0,
      b1[0], g1[0], be1[0], b2[0], g2[0], be2[0], yA, mA, n);

  mlp_layer_kernel<128, 128, true><<<mlpBlocks, blk, 0, stream>>>(
      nullptr, yA, (const float*)mA, cluster, W1_1, W2_1,
      b1[1], g1[1], be1[1], b2[1], g2[1], be2[1], yB, mB, n);

  zero_u32_kernel<<<(C * 64 + 255) / 256, blk, 0, stream>>>(mA, C * 64);

  mlp_layer_kernel<128, 128, true><<<mlpBlocks, blk, 0, stream>>>(
      nullptr, yB, (const float*)mB, cluster, W1_2, W2_2,
      b1[2], g1[2], be1[2], b2[2], g2[2], be2[2], yA, mA, n);

  zero_u32_kernel<<<(C * 64 + 255) / 256, blk, 0, stream>>>(mB, C * 64);

  final_linear_kernel<<<mlpBlocks, blk, 0, stream>>>(yA, (const float*)mA, cluster,
                                                     WLIN, linb, mB, n);

  norm_rows_kernel<<<(C + 7) / 8, blk, 0, stream>>>(mB, (float*)d_out, C);
}
